// CausalSelfAttention_66142496358995
// MI455X (gfx1250) — compile-verified, matrix operand order checked by reading, not run on hardware
//
#include <hip/hip_runtime.h>

typedef __attribute__((ext_vector_type(16))) __bf16 v16bf;
typedef __attribute__((ext_vector_type(8)))  float  v8f;

#define B_   2
#define T_   2048
#define D_   2048
#define H_   16
#define HD_  128
#define NTOK (B_ * T_)   // 4096
#define NQKV (3 * D_)    // 6144

__device__ __forceinline__ unsigned short f2bf(float f) {
    unsigned int u = __float_as_uint(f);
    u += 0x7FFFu + ((u >> 16) & 1u);     // round-to-nearest-even
    return (unsigned short)(u >> 16);
}
__device__ __forceinline__ float bf2f(unsigned short h) {
    return __uint_as_float(((unsigned int)h) << 16);
}

__device__ __forceinline__ v8f wmma_bf16(v16bf a, v16bf b, v8f c) {
    return __builtin_amdgcn_wmma_f32_16x16x32_bf16(false, a, false, b, (short)0, c,
                                                   false, false);
}

// Load a 16-bit A/B WMMA fragment from row-major memory.
// Caller passes p = base + row*ld + k0 + klo (klo = (lane>=16)?8:0).
// Layout: halves [k0..k0+7] -> VGPR0-3, halves [k0+16..k0+23] -> VGPR4-7.
__device__ __forceinline__ v16bf load_frag(const unsigned short* p) {
    v16bf f;
    uint4* d = reinterpret_cast<uint4*>(&f);
    d[0] = *reinterpret_cast<const uint4*>(p);
    d[1] = *reinterpret_cast<const uint4*>(p + 16);
    return f;
}

// ---------------------------------------------------------------- fp32 -> bf16
// Vectorized: float4 in (global_load_b128), packed 4xbf16 out (store_b64).
__global__ void cvt_bf16_kernel(const float* __restrict__ src,
                                unsigned short* __restrict__ dst, int n4) {
    int i = blockIdx.x * blockDim.x + threadIdx.x;
    if (i < n4) {
        float4 f = reinterpret_cast<const float4*>(src)[i];
        uint2 p;
        p.x = (unsigned)f2bf(f.x) | ((unsigned)f2bf(f.y) << 16);
        p.y = (unsigned)f2bf(f.z) | ((unsigned)f2bf(f.w) << 16);
        reinterpret_cast<uint2*>(dst)[i] = p;
    }
}

// ---------------------------------------------------------------- QKV GEMM
// C[tok, n] = sum_k x[tok,k] * qkv_w[n,k] + bias[n]  (NT GEMM, M=4096 N=6144 K=2048)
// 64x64 wave tile (4x4 accumulator tiles). Epilogue scatters into
// q/k [B,H,T,HD] (bf16) and vT [B,H,HD,T] (bf16).
__global__ __launch_bounds__(256) void gemm_qkv_kernel(
    const unsigned short* __restrict__ A,
    const unsigned short* __restrict__ W,
    const float* __restrict__ bias,
    unsigned short* __restrict__ qbuf,
    unsigned short* __restrict__ kbuf,
    unsigned short* __restrict__ vtbuf)
{
    const int lane = threadIdx.x & 31;
    const int ln   = lane & 15;
    const int hf   = lane >> 4;
    const int klo  = hf << 3;
    const int ntiles = NQKV / 64;                                   // 96
    const int wid = blockIdx.x * (blockDim.x >> 5) + (threadIdx.x >> 5);
    const int m0 = (wid / ntiles) * 64;
    const int n0 = (wid % ntiles) * 64;

    v8f acc[4][4] = {};
    const unsigned short* a0 = A + (size_t)(m0 + ln) * D_ + klo;
    const unsigned short* w0 = W + (size_t)(n0 + ln) * D_ + klo;

    for (int kk = 0; kk < D_; kk += 32) {
        v16bf a[4], b[4];
#pragma unroll
        for (int i = 0; i < 4; ++i) a[i] = load_frag(a0 + (size_t)i * 16 * D_ + kk);
#pragma unroll
        for (int j = 0; j < 4; ++j) b[j] = load_frag(w0 + (size_t)j * 16 * D_ + kk);
#pragma unroll
        for (int i = 0; i < 4; ++i)
#pragma unroll
            for (int j = 0; j < 4; ++j)
                acc[i][j] = wmma_bf16(a[i], b[j], acc[i][j]);
    }

    const int region = n0 >> 11;   // 0=Q 1=K 2=V (uniform per wave tile)
#pragma unroll
    for (int j = 0; j < 4; ++j) {
        const int n = n0 + j * 16 + ln;
        const float bv = bias[n];
        const int f = n & (D_ - 1);
        const int h = f >> 7;
        const int d = f & (HD_ - 1);
#pragma unroll
        for (int i = 0; i < 4; ++i) {
            const int rbase = m0 + i * 16 + hf * 8;    // token of C row v=0
            const int bidx  = rbase >> 11;             // batch (T_=2048)
            const int t0    = rbase & (T_ - 1);
            if (region == 2) {
                // V: store transposed -> vT[b,h,d,t]; 8 consecutive t per lane
                uint4 pk;
                unsigned int* pw = reinterpret_cast<unsigned int*>(&pk);
#pragma unroll
                for (int p = 0; p < 4; ++p) {
                    unsigned lo = f2bf(acc[i][j][2 * p]     + bv);
                    unsigned hi = f2bf(acc[i][j][2 * p + 1] + bv);
                    pw[p] = lo | (hi << 16);
                }
                *reinterpret_cast<uint4*>(
                    vtbuf + (((size_t)(bidx * H_ + h)) * HD_ + d) * T_ + t0) = pk;
            } else {
                unsigned short* dst = (region == 0) ? qbuf : kbuf;
#pragma unroll
                for (int v = 0; v < 8; ++v) {
                    const int t = t0 + v;
                    dst[(((size_t)(bidx * H_ + h)) * T_ + t) * HD_ + d] =
                        f2bf(acc[i][j][v] + bv);
                }
            }
        }
    }
}

// ---------------------------------------------------------------- RoPE (in place, bf16)
__global__ void rope_kernel(unsigned short* __restrict__ qk,
                            const float* __restrict__ cosb,
                            const float* __restrict__ sinb)
{
    const int i = blockIdx.x * blockDim.x + threadIdx.x;   // B*H*T*64 threads
    const int d = i & 63;
    const int t = (i >> 6) & (T_ - 1);
    const size_t base = (size_t)(i >> 6) * HD_;
    const float x1 = bf2f(qk[base + d]);
    const float x2 = bf2f(qk[base + d + 64]);
    const float c1 = cosb[t * HD_ + d],      s1 = sinb[t * HD_ + d];
    const float c2 = cosb[t * HD_ + d + 64], s2 = sinb[t * HD_ + d + 64];
    qk[base + d]      = f2bf(x1 * c1 - x2 * s1);   // x*cos + (-x2)*sin
    qk[base + d + 64] = f2bf(x2 * c2 + x1 * s2);   // x*cos + ( x1)*sin
}

// ---------------------------------------------------------------- flash attention
// One wave per (bh, 16-query tile). Keys processed 32 at a time.
__global__ __launch_bounds__(32) void attn_kernel(
    const unsigned short* __restrict__ qbuf,
    const unsigned short* __restrict__ kbuf,
    const unsigned short* __restrict__ vtbuf,
    unsigned short* __restrict__ ob)           // [NTOK, D_] bf16
{
    __shared__ unsigned short lp[16 * 32];     // P tile, row-major 16x32 bf16
    const int qt   = blockIdx.x;               // 0..127
    const int bh   = blockIdx.y;               // 0..31
    const int lane = threadIdx.x;
    const int ln   = lane & 15;
    const int hf   = lane >> 4;
    const int klo  = hf << 3;

    const unsigned short* Q = qbuf  + (size_t)bh * T_ * HD_;
    const unsigned short* K = kbuf  + (size_t)bh * T_ * HD_;
    const unsigned short* V = vtbuf + (size_t)bh * HD_ * T_;   // [HD, T]

    v16bf qf[4];
#pragma unroll
    for (int s = 0; s < 4; ++s)
        qf[s] = load_frag(Q + (size_t)(qt * 16 + ln) * HD_ + s * 32 + klo);

    v8f O[8] = {};
    float mr[8], lr[8];
#pragma unroll
    for (int v = 0; v < 8; ++v) { mr[v] = -1e30f; lr[v] = 0.0f; }

    const int kbmax = (qt * 16 + 15) >> 5;
    const float sc = 0.08838834764831845f;     // 1/sqrt(128)

    for (int kbk = 0; kbk <= kbmax; ++kbk) {
        // S = Q * K^T : 16 queries x 32 keys = two C tiles
        v8f s0 = {}, s1 = {};
#pragma unroll
        for (int s = 0; s < 4; ++s) {
            v16bf b0 = load_frag(K + (size_t)(kbk * 32 + ln)      * HD_ + s * 32 + klo);
            v16bf b1 = load_frag(K + (size_t)(kbk * 32 + 16 + ln) * HD_ + s * 32 + klo);
            s0 = wmma_bf16(qf[s], b0, s0);
            s1 = wmma_bf16(qf[s], b1, s1);
        }
        const bool diag = (kbk * 32 + 31 > qt * 16);
        float p0[8], p1[8], t8[8];
#pragma unroll
        for (int v = 0; v < 8; ++v) {
            float e0 = s0[v] * sc, e1 = s1[v] * sc;
            if (diag) {
                const int r = qt * 16 + hf * 8 + v;
                if (kbk * 32 + ln      > r) e0 = -1e30f;
                if (kbk * 32 + 16 + ln > r) e1 = -1e30f;
            }
            p0[v] = e0; p1[v] = e1;
            t8[v] = fmaxf(e0, e1);
        }
        // row max across the 16 lanes of each half
#pragma unroll
        for (int v = 0; v < 8; ++v) {
#pragma unroll
            for (int off = 1; off < 16; off <<= 1)
                t8[v] = fmaxf(t8[v], __shfl_xor(t8[v], off, 32));
        }
        float alpha[8];
#pragma unroll
        for (int v = 0; v < 8; ++v) {
            const float mnew = fmaxf(mr[v], t8[v]);
            alpha[v] = __expf(mr[v] - mnew);
            mr[v] = mnew;
            p0[v] = __expf(p0[v] - mnew);
            p1[v] = __expf(p1[v] - mnew);
            t8[v] = p0[v] + p1[v];
        }
#pragma unroll
        for (int v = 0; v < 8; ++v) {
#pragma unroll
            for (int off = 1; off < 16; off <<= 1)
                t8[v] += __shfl_xor(t8[v], off, 32);
            lr[v] = lr[v] * alpha[v] + t8[v];
        }
#pragma unroll
        for (int n = 0; n < 8; ++n)
#pragma unroll
            for (int v = 0; v < 8; ++v)
                O[n][v] *= alpha[v];

        // C-layout P -> LDS row-major, reload as A fragment
#pragma unroll
        for (int v = 0; v < 8; ++v) {
            lp[(hf * 8 + v) * 32 + ln]      = f2bf(p0[v]);
            lp[(hf * 8 + v) * 32 + ln + 16] = f2bf(p1[v]);
        }
        asm volatile("s_wait_dscnt 0" ::: "memory");
        v16bf pf;
        {
            uint4* pd = reinterpret_cast<uint4*>(&pf);
            pd[0] = *reinterpret_cast<const uint4*>(lp + ln * 32 + klo);
            pd[1] = *reinterpret_cast<const uint4*>(lp + ln * 32 + klo + 16);
        }
        // O += P * V  (B operand from transposed V: rows = d, K = keys)
#pragma unroll
        for (int n = 0; n < 8; ++n) {
            v16bf vf = load_frag(V + (size_t)(n * 16 + ln) * T_ + kbk * 32 + klo);
            O[n] = wmma_bf16(pf, vf, O[n]);
        }
        asm volatile("s_wait_dscnt 0" ::: "memory");  // WAR fence before next P store
    }

    const int b = bh >> 4;
    const int h = bh & 15;
#pragma unroll
    for (int v = 0; v < 8; ++v) {
        const float inv = 1.0f / lr[v];
        const int t = qt * 16 + hf * 8 + v;
        const size_t row = ((size_t)(b * T_ + t)) * D_ + h * HD_;
#pragma unroll
        for (int n = 0; n < 8; ++n)
            ob[row + n * 16 + ln] = f2bf(O[n][v] * inv);
    }
}

// ---------------------------------------------------------------- output GEMM
// out[tok, n] = sum_k attn[tok,k] * out_w[n,k] + out_b[n]   (fp32 result)
// 64x64 wave tile (4x4 accumulator tiles).
__global__ __launch_bounds__(256) void gemm_out_kernel(
    const unsigned short* __restrict__ A,
    const unsigned short* __restrict__ W,
    const float* __restrict__ bias,
    float* __restrict__ out)
{
    const int lane = threadIdx.x & 31;
    const int ln   = lane & 15;
    const int hf   = lane >> 4;
    const int klo  = hf << 3;
    const int ntiles = D_ / 64;                                     // 32
    const int wid = blockIdx.x * (blockDim.x >> 5) + (threadIdx.x >> 5);
    const int m0 = (wid / ntiles) * 64;
    const int n0 = (wid % ntiles) * 64;

    v8f acc[4][4] = {};
    const unsigned short* a0 = A + (size_t)(m0 + ln) * D_ + klo;
    const unsigned short* w0 = W + (size_t)(n0 + ln) * D_ + klo;

    for (int kk = 0; kk < D_; kk += 32) {
        v16bf a[4], b[4];
#pragma unroll
        for (int i = 0; i < 4; ++i) a[i] = load_frag(a0 + (size_t)i * 16 * D_ + kk);
#pragma unroll
        for (int j = 0; j < 4; ++j) b[j] = load_frag(w0 + (size_t)j * 16 * D_ + kk);
#pragma unroll
        for (int i = 0; i < 4; ++i)
#pragma unroll
            for (int j = 0; j < 4; ++j)
                acc[i][j] = wmma_bf16(a[i], b[j], acc[i][j]);
    }

#pragma unroll
    for (int j = 0; j < 4; ++j) {
        const int n = n0 + j * 16 + ln;
        const float bv = bias[n];
#pragma unroll
        for (int i = 0; i < 4; ++i) {
            const int rbase = m0 + i * 16 + hf * 8;
#pragma unroll
            for (int v = 0; v < 8; ++v)
                out[(size_t)(rbase + v) * D_ + n] = acc[i][j][v] + bv;
        }
    }
}

// ---------------------------------------------------------------- launch
extern "C" void kernel_launch(void* const* d_in, const int* in_sizes, int n_in,
                              void* d_out, int out_size, void* d_ws, size_t ws_size,
                              hipStream_t stream) {
    const float* x        = (const float*)d_in[0];
    const float* rope_cos = (const float*)d_in[1];
    const float* rope_sin = (const float*)d_in[2];
    const float* qkv_w    = (const float*)d_in[3];
    const float* qkv_b    = (const float*)d_in[4];
    const float* out_w    = (const float*)d_in[5];
    const float* out_b    = (const float*)d_in[6];
    float* out = (float*)d_out;

    unsigned short* xb    = (unsigned short*)d_ws;                 // [4096,2048]
    unsigned short* wqkv  = xb    + (size_t)NTOK * D_;             // [6144,2048]
    unsigned short* wout  = wqkv  + (size_t)NQKV * D_;             // [2048,2048]
    unsigned short* qbuf  = wout  + (size_t)D_ * D_;               // [B,H,T,HD]
    unsigned short* kbuf  = qbuf  + (size_t)B_ * H_ * T_ * HD_;    // [B,H,T,HD]
    unsigned short* vtbuf = kbuf  + (size_t)B_ * H_ * T_ * HD_;    // [B,H,HD,T]
    unsigned short* ob    = vtbuf + (size_t)B_ * H_ * T_ * HD_;    // [4096,2048]

    const int n_x  = NTOK * D_;       // 8388608
    const int n_wq = NQKV * D_;       // 12582912
    const int n_wo = D_ * D_;         // 4194304

    cvt_bf16_kernel<<<(n_x  / 4 + 255) / 256, 256, 0, stream>>>(x,     xb,   n_x  / 4);
    cvt_bf16_kernel<<<(n_wq / 4 + 255) / 256, 256, 0, stream>>>(qkv_w, wqkv, n_wq / 4);
    cvt_bf16_kernel<<<(n_wo / 4 + 255) / 256, 256, 0, stream>>>(out_w, wout, n_wo / 4);

    // 64 M-tiles x 96 N-tiles = 6144 waves, 8 waves/block
    gemm_qkv_kernel<<<6144 / 8, 256, 0, stream>>>(xb, wqkv, qkv_b,
                                                  qbuf, kbuf, vtbuf);

    const int n_rope = B_ * H_ * T_ * 64;   // 4194304 threads
    rope_kernel<<<n_rope / 256, 256, 0, stream>>>(qbuf, rope_cos, rope_sin);
    rope_kernel<<<n_rope / 256, 256, 0, stream>>>(kbuf, rope_cos, rope_sin);

    attn_kernel<<<dim3(T_ / 16, B_ * H_), 32, 0, stream>>>(qbuf, kbuf, vtbuf, ob);

    // 64 M-tiles x 32 N-tiles = 2048 waves, 8 waves/block
    gemm_out_kernel<<<2048 / 8, 256, 0, stream>>>(ob, wout, out_b, out);
}
